// GIN_64158221467926
// MI455X (gfx1250) — matrix operand order repaired and checked by reading, not compiled, not run on hardware
//
#include <hip/hip_runtime.h>

#define NNODES   50000
#define NEDGES   800000
#define HID      128
#define NGRAPHS  64
#define NCLS     16
#define MTILES   (NNODES / 16)   // 3125 exactly

typedef __attribute__((ext_vector_type(16))) __bf16    v16bf;
typedef __attribute__((ext_vector_type(8)))  float     v8f;

// ---------------- weight prep: W[K,N] f32 -> Wt[N,K] bf16 (transpose) -------
__global__ __launch_bounds__(256) void prep_weight(const float* __restrict__ W,
                                                   unsigned short* __restrict__ Wt) {
    int idx = blockIdx.x * 256 + threadIdx.x;       // 0 .. 16383
    if (idx < HID * HID) {
        int k = idx >> 7, n = idx & 127;
        __bf16 v = (__bf16)W[k * HID + n];          // native f32->bf16 cvt
        Wt[n * HID + k] = __builtin_bit_cast(unsigned short, v);
    }
}

// ---------------- generic helpers ------------------------------------------
__global__ __launch_bounds__(256) void zero_f32(float* __restrict__ p, int n) {
    int t = blockIdx.x * 256 + threadIdx.x;
    if (t < n) p[t] = 0.0f;
}

__global__ __launch_bounds__(256) void copy_f32_v4(const float* __restrict__ in,
                                                   float* __restrict__ out, int n4) {
    int t = blockIdx.x * 256 + threadIdx.x;
    if (t < n4) ((float4*)out)[t] = ((const float4*)in)[t];
}

// ---------------- edge scatter-add: Aggr[dst] += X[src] --------------------
// 32 lanes per edge, 4 floats per lane (float4 gather, 4 f32 atomics).
__global__ __launch_bounds__(256) void scatter_edges(const int* __restrict__ src,
                                                     const int* __restrict__ dst,
                                                     const float* __restrict__ X,
                                                     float* __restrict__ Aggr,
                                                     int nEdges) {
    unsigned t = blockIdx.x * 256u + threadIdx.x;
    int e = (int)(t >> 5);
    int lane = (int)(t & 31u);
    if (e >= nEdges) return;
    int s = src[e], d = dst[e];
    float4 v = *(const float4*)(X + (size_t)s * HID + lane * 4);
    float* o = Aggr + (size_t)d * HID + lane * 4;
    atomicAdd(o + 0, v.x);
    atomicAdd(o + 1, v.y);
    atomicAdd(o + 2, v.z);
    atomicAdd(o + 3, v.w);
}

// ---------------- WMMA GEMM: Out = relu(Ain[M,128] @ W[128,128] + bias) ----
// Ain f32 (converted to bf16 fragments via native cvt), W pre-transposed bf16
// Wt[N,K]. One wave per 16-row M tile; 8 N tiles x 4 K steps = 32 WMMAs.
__global__ __launch_bounds__(256) void gemm_bias_relu(const float* __restrict__ Ain,
                                                      const unsigned short* __restrict__ Wt,
                                                      const float* __restrict__ bias,
                                                      float* __restrict__ Out,
                                                      int mTiles) {
    int lane   = threadIdx.x & 31;
    int wave   = threadIdx.x >> 5;
    int tile   = blockIdx.x * 8 + wave;     // wave-uniform -> EXEC stays full
    if (tile >= mTiles) return;
    int half   = lane >> 4;                 // 0 or 1
    int lane16 = lane & 15;
    int arow   = tile * 16 + lane16;        // A-matrix row for this lane

    // ---- A fragments: contiguous k order, same as B.
    // Element s holds K = half*16 + s (+ kb*32): 16 contiguous floats at base.
    v16bf afrag[4];
    const float* ap = Ain + (size_t)arow * HID;
#pragma unroll
    for (int kb = 0; kb < 4; ++kb) {
        int base = kb * 32 + half * 16;
        float4 q0 = *(const float4*)(ap + base);
        float4 q1 = *(const float4*)(ap + base + 4);
        float4 q2 = *(const float4*)(ap + base + 8);
        float4 q3 = *(const float4*)(ap + base + 12);
        v16bf a;
        a[0]  = (__bf16)q0.x;  a[1]  = (__bf16)q0.y;
        a[2]  = (__bf16)q0.z;  a[3]  = (__bf16)q0.w;
        a[4]  = (__bf16)q1.x;  a[5]  = (__bf16)q1.y;
        a[6]  = (__bf16)q1.z;  a[7]  = (__bf16)q1.w;
        a[8]  = (__bf16)q2.x;  a[9]  = (__bf16)q2.y;
        a[10] = (__bf16)q2.z;  a[11] = (__bf16)q2.w;
        a[12] = (__bf16)q3.x;  a[13] = (__bf16)q3.y;
        a[14] = (__bf16)q3.z;  a[15] = (__bf16)q3.w;
        afrag[kb] = a;
    }

    // ---- loop over 8 N tiles, accumulate K=128 in 4 WMMA steps
#pragma unroll
    for (int n = 0; n < 8; ++n) {
        int ncol = n * 16 + lane16;         // B column owned by this lane
        const unsigned short* wp = Wt + (size_t)ncol * HID;
        v8f c = {};
#pragma unroll
        for (int kb = 0; kb < 4; ++kb) {
            // B in the same contiguous k order as A: pair j holds K = half*16 + 2j (+ kb*32)
            // -> 16 contiguous bf16 = one 32B run, loads as 2x b128.
            v16bf bfrag = *(const v16bf*)(wp + kb * 32 + half * 16);
            c = __builtin_amdgcn_wmma_f32_16x16x32_bf16(
                    false, afrag[kb], false, bfrag, (short)0, c, false, false);
        }
        float bv = bias[ncol];
#pragma unroll
        for (int r = 0; r < 8; ++r) {       // C/D: row = half*8 + r, col = lane16
            int orow = tile * 16 + half * 8 + r;
            float v = c[r] + bv;
            Out[(size_t)orow * HID + ncol] = v > 0.0f ? v : 0.0f;
        }
    }
}

// ---------------- BatchNorm: stats, finalize, apply ------------------------
__global__ __launch_bounds__(128) void bn_stats(const float* __restrict__ H,
                                                float* __restrict__ sums,
                                                float* __restrict__ sumsq, int nRows) {
    int f = threadIdx.x;                    // feature 0..127
    float s = 0.0f, s2 = 0.0f;
    for (int r = blockIdx.x; r < nRows; r += gridDim.x) {
        float v = H[(size_t)r * HID + f];
        s += v; s2 += v * v;
    }
    atomicAdd(&sums[f], s);
    atomicAdd(&sumsq[f], s2);
}

__global__ __launch_bounds__(128) void bn_finalize(const float* __restrict__ sums,
                                                   const float* __restrict__ sumsq,
                                                   const float* __restrict__ gamma,
                                                   const float* __restrict__ beta,
                                                   float* __restrict__ scale,
                                                   float* __restrict__ shift, int nRows) {
    int f = threadIdx.x;
    float inv = 1.0f / (float)nRows;
    float mu  = sums[f] * inv;
    float var = sumsq[f] * inv - mu * mu;
    float sc  = gamma[f] * rsqrtf(var + 1e-5f);
    scale[f] = sc;
    shift[f] = beta[f] - mu * sc;
}

// writes X (next-layer input) and initializes Aggr = X for the next scatter
__global__ __launch_bounds__(256) void bn_apply(const float* __restrict__ H,
                                                const float* __restrict__ scale,
                                                const float* __restrict__ shift,
                                                float* __restrict__ X,
                                                float* __restrict__ Aggr, int n) {
    int t = blockIdx.x * 256 + threadIdx.x;
    if (t < n) {
        int f = t & 127;
        float v = H[t] * scale[f] + shift[f];
        X[t] = v;
        Aggr[t] = v;
    }
}

// ---------------- global mean pool + classifier head -----------------------
__global__ __launch_bounds__(128) void pool_accum(const float* __restrict__ X,
                                                  const int* __restrict__ batch,
                                                  float* __restrict__ gsum,
                                                  float* __restrict__ gcnt, int nRows) {
    int f = threadIdx.x;
    for (int r = blockIdx.x; r < nRows; r += gridDim.x) {
        int g = batch[r];
        atomicAdd(&gsum[g * HID + f], X[(size_t)r * HID + f]);
        if (f == 0) atomicAdd(&gcnt[g], 1.0f);
    }
}

__global__ __launch_bounds__(256) void fc1_kernel(const float* __restrict__ gsum,
                                                  const float* __restrict__ gcnt,
                                                  const float* __restrict__ W,
                                                  const float* __restrict__ b,
                                                  float* __restrict__ out) {
    int idx = blockIdx.x * 256 + threadIdx.x;
    if (idx >= NGRAPHS * HID) return;
    int g = idx >> 7, f = idx & 127;
    float inv = 1.0f / fmaxf(gcnt[g], 1.0f);
    float acc = 0.0f;
    const float* gr = gsum + g * HID;
#pragma unroll 8
    for (int k = 0; k < HID; ++k) acc += gr[k] * W[k * HID + f];
    acc = acc * inv + b[f];
    out[idx] = fmaxf(acc, 0.0f);
}

__global__ __launch_bounds__(64) void head_kernel(const float* __restrict__ fc1out,
                                                  const float* __restrict__ W,
                                                  const float* __restrict__ b,
                                                  float* __restrict__ out) {
    int g = threadIdx.x;                    // 0..63, one graph per lane
    float logit[NCLS];
#pragma unroll
    for (int c = 0; c < NCLS; ++c) {
        float acc = b[c];
        for (int k = 0; k < HID; ++k) acc += fc1out[g * HID + k] * W[k * NCLS + c];
        logit[c] = acc;
    }
    float m = logit[0];
#pragma unroll
    for (int c = 1; c < NCLS; ++c) m = fmaxf(m, logit[c]);
    float s = 0.0f;
#pragma unroll
    for (int c = 0; c < NCLS; ++c) s += expf(logit[c] - m);
    float lse = m + logf(s);
#pragma unroll
    for (int c = 0; c < NCLS; ++c) out[g * NCLS + c] = logit[c] - lse;
}

// ---------------------------------------------------------------------------
extern "C" void kernel_launch(void* const* d_in, const int* in_sizes, int n_in,
                              void* d_out, int out_size, void* d_ws, size_t ws_size,
                              hipStream_t stream) {
    const float* x_in  = (const float*)d_in[0];
    const int*   src   = (const int*)d_in[1];
    const int*   dst   = src + NEDGES;
    const int*   batch = (const int*)d_in[2];
    const float *W1[4], *b1[4], *W2[4], *b2[4], *bng[4], *bnb[4];
    for (int l = 0; l < 4; ++l) {
        W1[l]  = (const float*)d_in[3 + l * 6 + 0];
        b1[l]  = (const float*)d_in[3 + l * 6 + 1];
        W2[l]  = (const float*)d_in[3 + l * 6 + 2];
        b2[l]  = (const float*)d_in[3 + l * 6 + 3];
        bng[l] = (const float*)d_in[3 + l * 6 + 4];
        bnb[l] = (const float*)d_in[3 + l * 6 + 5];
    }
    const float* fc1W = (const float*)d_in[27];
    const float* fc1b = (const float*)d_in[28];
    const float* fc2W = (const float*)d_in[29];
    const float* fc2b = (const float*)d_in[30];

    // ---- carve workspace (256B aligned chunks) ----
    char*  ws  = (char*)d_ws;
    size_t off = 0;
    auto carve = [&](size_t bytes) -> char* {
        char* p = ws + off;
        off = (off + bytes + 255) & ~(size_t)255;
        return p;
    };
    float*          A     = (float*)carve(sizeof(float) * NNODES * HID);  // aggregation
    float*          H1    = (float*)carve(sizeof(float) * NNODES * HID);  // also "X"
    float*          H2    = (float*)carve(sizeof(float) * NNODES * HID);
    unsigned short* Wt    = (unsigned short*)carve(sizeof(unsigned short) * 8 * HID * HID);
    float*          bnsum = (float*)carve(sizeof(float) * HID);   // contiguous with bnsq
    float*          bnsq  = (float*)carve(sizeof(float) * HID);
    float*          bnsc  = (float*)carve(sizeof(float) * HID);
    float*          bnsh  = (float*)carve(sizeof(float) * HID);
    float*          gsum  = (float*)carve(sizeof(float) * NGRAPHS * HID); // contiguous w/ gcnt
    float*          gcnt  = (float*)carve(sizeof(float) * NGRAPHS);
    float*          f1o   = (float*)carve(sizeof(float) * NGRAPHS * HID);
    (void)in_sizes; (void)n_in; (void)out_size; (void)ws_size;

    // ---- convert+transpose all GEMM weights to bf16 once ----
    for (int l = 0; l < 4; ++l) {
        prep_weight<<<64, 256, 0, stream>>>(W1[l], Wt + (size_t)(2 * l + 0) * HID * HID);
        prep_weight<<<64, 256, 0, stream>>>(W2[l], Wt + (size_t)(2 * l + 1) * HID * HID);
    }

    // ---- A = x (GIN eps=0: h = x + sum_neighbors) ----
    copy_f32_v4<<<(NNODES * HID / 4 + 255) / 256, 256, 0, stream>>>(x_in, A, NNODES * HID / 4);

    for (int l = 0; l < 4; ++l) {
        const float* X = (l == 0) ? x_in : H1;                 // H1 doubles as X
        scatter_edges<<<(NEDGES * 32) / 256, 256, 0, stream>>>(src, dst, X, A, NEDGES);
        gemm_bias_relu<<<(MTILES + 7) / 8, 256, 0, stream>>>(
            A, Wt + (size_t)(2 * l + 0) * HID * HID, b1[l], H1, MTILES);
        gemm_bias_relu<<<(MTILES + 7) / 8, 256, 0, stream>>>(
            H1, Wt + (size_t)(2 * l + 1) * HID * HID, b2[l], H2, MTILES);
        zero_f32<<<1, 256, 0, stream>>>(bnsum, 2 * HID);       // bnsum+bnsq contiguous
        bn_stats<<<256, 128, 0, stream>>>(H2, bnsum, bnsq, NNODES);
        bn_finalize<<<1, 128, 0, stream>>>(bnsum, bnsq, bng[l], bnb[l], bnsc, bnsh, NNODES);
        bn_apply<<<(NNODES * HID + 255) / 256, 256, 0, stream>>>(H2, bnsc, bnsh, H1, A,
                                                                 NNODES * HID);
    }

    // ---- mean pool + heads ----
    zero_f32<<<(NGRAPHS * HID + NGRAPHS + 255) / 256, 256, 0, stream>>>(
        gsum, NGRAPHS * HID + NGRAPHS);                        // gsum+gcnt contiguous
    pool_accum<<<512, 128, 0, stream>>>(H1, batch, gsum, gcnt, NNODES);
    fc1_kernel<<<(NGRAPHS * HID + 255) / 256, 256, 0, stream>>>(gsum, gcnt, fc1W, fc1b, f1o);
    head_kernel<<<1, 64, 0, stream>>>(f1o, fc2W, fc2b, (float*)d_out);
}
